// MultiHeadAttention_12902081757925
// MI455X (gfx1250) — matrix operand order repaired and checked by reading, not compiled, not run on hardware
//
#include <hip/hip_runtime.h>
#include <hip/hip_bf16.h>

// ---------------------------------------------------------------------------
// CDNA5 (gfx1250) fused multi-head attention, bf16 WMMA pipeline.
// Shapes: SEQ=4096, IN=256, H=8, D=32, D_MODEL=256.
// d_out = [ normed 4096x256 f32 | attn 8x4096x4096 f32 ]
//
// v2: scores + softmax + context fully fused. A block owns 16 q-rows of one
// head; the whole 16x4096 f32 score stripe lives in LDS (263KB of the 320KB
// WGP LDS), so the 536MB attn tensor is written to HBM exactly once
// (normalized) -- the roofline minimum for this problem.
// ---------------------------------------------------------------------------

typedef __bf16 v16bf __attribute__((ext_vector_type(16)));
typedef float  v8f   __attribute__((ext_vector_type(8)));

#define SEQ    4096
#define DMODEL 256
#define NHEADS 8
#define DHEAD  32

// ---- WMMA helpers (CDNA5 ISA 7.12.2 layouts, wave32) ----------------------

__device__ __forceinline__ v8f wmma_bf16(v16bf a, v16bf b, v8f c) {
  // D = A(16x32 bf16) x B(32x16 bf16) + C(16x16 f32)
  return __builtin_amdgcn_wmma_f32_16x16x32_bf16(
      /*neg_a=*/false, a, /*neg_b=*/false, b,
      /*c_mod=*/(short)0, c, /*reuse_a=*/false, /*reuse_b=*/false);
}

// A-matrix 16x32 (MxK), 16-bit: lane L holds row M=L&15; contiguous K:
// lane half h holds K=16h..16h+15 (matches B fragment loaders).
__device__ __forceinline__ v16bf load_frag_a(const __bf16* base, int ldk) {
  const int lane = threadIdx.x & 31;
  const int row  = lane & 15;
  const int half = lane >> 4;
  v16bf a;
#pragma unroll
  for (int v = 0; v < 8; ++v) {
    const int k = 2 * v + 16 * half;
    a[2 * v]     = base[row * ldk + k];
    a[2 * v + 1] = base[row * ldk + k + 1];
  }
  return a;
}

// Same A layout but sourcing f32 (LDS score stripe), converting to bf16.
__device__ __forceinline__ v16bf load_frag_a_f32(const float* base, int ldk) {
  const int lane = threadIdx.x & 31;
  const int row  = lane & 15;
  const int half = lane >> 4;
  v16bf a;
#pragma unroll
  for (int v = 0; v < 8; ++v) {
    const int k = 2 * v + 16 * half;
    a[2 * v]     = (__bf16)base[row * ldk + k];
    a[2 * v + 1] = (__bf16)base[row * ldk + k + 1];
  }
  return a;
}

// B-matrix 32x16 (KxN) from an [N][K] (transposed) tile: B(k,n)=tile[n][k].
__device__ __forceinline__ v16bf load_frag_b_nk(const __bf16* base, int ldk) {
  const int lane = threadIdx.x & 31;
  const int col  = lane & 15;
  const int kb   = (lane >> 4) * 16;
  v16bf b;
#pragma unroll
  for (int v = 0; v < 8; ++v) {
    b[2 * v]     = base[(size_t)col * ldk + kb + 2 * v];
    b[2 * v + 1] = base[(size_t)col * ldk + kb + 2 * v + 1];
  }
  return b;
}

// B-matrix 32x16 (KxN) from a row-major [K][N] source.
__device__ __forceinline__ v16bf load_frag_b_kn(const __bf16* base, int ldn) {
  const int lane = threadIdx.x & 31;
  const int col  = lane & 15;
  const int kb   = (lane >> 4) * 16;
  v16bf b;
#pragma unroll
  for (int v = 0; v < 8; ++v) {
    b[2 * v]     = base[(size_t)(kb + 2 * v) * ldn + col];
    b[2 * v + 1] = base[(size_t)(kb + 2 * v + 1) * ldn + col];
  }
  return b;
}

// ---- Kernel 1: f32 -> bf16 ------------------------------------------------

__global__ void __launch_bounds__(256) cvt_f32_bf16(const float* __restrict__ src,
                                                    __bf16* __restrict__ dst,
                                                    int nelem) {
  int i = blockIdx.x * 256 + threadIdx.x;
  const int stride = gridDim.x * 256;
  for (; i < nelem; i += stride) dst[i] = (__bf16)src[i];
}

// ---- Kernel 2: Out[MxN] = X[MxK] @ W[NxK]^T  (all bf16, f32 accum) --------
// Block tile 64x64, 256 threads = 8 waves, each wave 2 adjacent 16x16 tiles.

__global__ void __launch_bounds__(256) gemm_xwt_bf16(const __bf16* __restrict__ X,
                                                     const __bf16* __restrict__ W,
                                                     __bf16* __restrict__ Out,
                                                     int M, int N, int Kf) {
  __shared__ __bf16 As[64][40];  // padded stride: 80B/row avoids bank clash
  __shared__ __bf16 Bs[64][40];
  const int m0 = blockIdx.y * 64, n0 = blockIdx.x * 64;
  const int tid = threadIdx.x;
  const int w = tid >> 5;
  const int tm = w >> 1;          // wave's M-tile (0..3)
  const int tn2 = (w & 1) * 2;    // wave's first N-tile (0 or 2)
  const int r = tid >> 2, c8 = (tid & 3) * 8;

  v8f acc0 = {}, acc1 = {};
  for (int kc = 0; kc < Kf; kc += 32) {
#pragma unroll
    for (int e = 0; e < 8; ++e) {
      As[r][c8 + e] = X[(size_t)(m0 + r) * Kf + kc + c8 + e];
      Bs[r][c8 + e] = W[(size_t)(n0 + r) * Kf + kc + c8 + e];
    }
    if (kc + 32 < Kf) {
      __builtin_prefetch(&X[(size_t)(m0 + r) * Kf + kc + 32 + c8], 0, 1);
      __builtin_prefetch(&W[(size_t)(n0 + r) * Kf + kc + 32 + c8], 0, 1);
    }
    __syncthreads();
    v16bf a  = load_frag_a(&As[tm * 16][0], 40);
    v16bf b0 = load_frag_b_nk(&Bs[tn2 * 16][0], 40);
    v16bf b1 = load_frag_b_nk(&Bs[(tn2 + 1) * 16][0], 40);
    acc0 = wmma_bf16(a, b0, acc0);
    acc1 = wmma_bf16(a, b1, acc1);
    __syncthreads();
  }
  const int lane = tid & 31, col = lane & 15, half = lane >> 4;
#pragma unroll
  for (int rr = 0; rr < 8; ++rr) {
    const int m = m0 + tm * 16 + rr + 8 * half;
    Out[(size_t)m * N + n0 + tn2 * 16 + col]       = (__bf16)acc0[rr];
    Out[(size_t)m * N + n0 + (tn2 + 1) * 16 + col] = (__bf16)acc1[rr];
  }
}

// ---- Kernel 3: fused scores + mask + softmax + context --------------------
// One block = 16 q-rows of one head, 256 threads = 8 waves.
//   phase 1: S[16 x 4096] = Q_h K_h^T * scale (WMMA; A-fragment loaded once,
//            reused across all 256 j-tiles; K_h is 256KB -> L2 resident)
//   phase 2: softmax entirely inside the LDS stripe (mask applied in phase 1)
//   phase 3: single normalized write of attn to HBM (roofline minimum)
//   phase 4: context = P @ V_h (WMMA; A from LDS f32 stripe, cvt to bf16;
//            8 waves split the 128 K-chunks, merged with ds_add_f32)

__global__ void __launch_bounds__(256) fused_attn_kernel(
    const __bf16* __restrict__ Qb, const __bf16* __restrict__ Kb,
    const __bf16* __restrict__ Vb, const unsigned char* __restrict__ mask,
    float* __restrict__ attn,      // [H,n,n] output probabilities
    __bf16* __restrict__ Ctx,      // [n, DMODEL] bf16 out
    int n) {
  __shared__ float  Sf[16][4104];  // 263KB stripe; pad 8 -> skewed banks
  __shared__ __bf16 Qs[16][40];
  __shared__ float  red[16][16];
  __shared__ float  rowmax[16], rowinv[16];
  __shared__ float  ctxacc[16][32];

  const int h   = blockIdx.y;
  const int q0  = blockIdx.x * 16;
  const int tid = threadIdx.x;
  const int w   = tid >> 5;
  const int lane = tid & 31, col = lane & 15, half = lane >> 4;

  // stage Q tile (16x32 bf16) and zero the context accumulator
  if (tid < 128) {
    const int r = tid >> 3, c4 = (tid & 7) * 4;
#pragma unroll
    for (int e = 0; e < 4; ++e)
      Qs[r][c4 + e] = Qb[(size_t)(q0 + r) * DMODEL + h * DHEAD + c4 + e];
  }
  ((float*)ctxacc)[tid]       = 0.f;
  ((float*)ctxacc)[tid + 256] = 0.f;
  __syncthreads();

  // ---- phase 1: scores into LDS -----------------------------------------
  const float scale = 0.17677669529663687f;  // 1/sqrt(32)
  v16bf aq = load_frag_a(&Qs[0][0], 40);     // reused for all j-tiles
  for (int t = w * 32; t < (w + 1) * 32; ++t) {
    const int j0 = t * 16;
    v16bf b = load_frag_b_nk(Kb + (size_t)j0 * DMODEL + h * DHEAD, DMODEL);
    v8f acc = {};
    acc = wmma_bf16(aq, b, acc);
#pragma unroll
    for (int rr = 0; rr < 8; ++rr) {
      const int m = rr + 8 * half;
      const int j = j0 + col;
      float s = acc[rr] * scale;
      if (mask[(size_t)(q0 + m) * n + j]) s = -1e9f;
      Sf[m][j] = s;
    }
  }
  __syncthreads();

  // ---- phase 2: softmax in LDS -------------------------------------------
  const int r = tid >> 4;   // q-row 0..15
  const int l = tid & 15;   // 16 lanes per row
  float mx = -1e30f;
  for (int j = l; j < n; j += 16) mx = fmaxf(mx, Sf[r][j]);
  red[r][l] = mx;
  __syncthreads();
  if (l == 0) {
    float mm = red[r][0];
#pragma unroll
    for (int i = 1; i < 16; ++i) mm = fmaxf(mm, red[r][i]);
    rowmax[r] = mm;
  }
  __syncthreads();
  const float mr = rowmax[r];
  float s = 0.f;
  for (int j = l; j < n; j += 16) {
    const float e = __expf(Sf[r][j] - mr);
    Sf[r][j] = e;              // keep unnormalized exp for context GEMM
    s += e;
  }
  red[r][l] = s;
  __syncthreads();
  if (l == 0) {
    float ss = red[r][0];
#pragma unroll
    for (int i = 1; i < 16; ++i) ss += red[r][i];
    rowinv[r] = 1.0f / ss;
  }
  __syncthreads();

  // ---- phase 3: single normalized attn write to HBM ----------------------
  const float inv = rowinv[r];
  float* Sg = attn + (size_t)h * n * n + (size_t)q0 * n;
  for (int j = l; j < n; j += 16)
    Sg[(size_t)r * n + j] = Sf[r][j] * inv;

  // ---- phase 4: context = P @ V_h (1/rowsum folded into epilogue) --------
  v8f acc0 = {}, acc1 = {};
  for (int c = w; c < n / 32; c += 8) {          // 16 K-chunks per wave
    v16bf a = load_frag_a_f32(&Sf[0][c * 32], 4104);
    const __bf16* Vbase = Vb + (size_t)(c * 32) * DMODEL + h * DHEAD;
    v16bf b0 = load_frag_b_kn(Vbase, DMODEL);       // d = 0..15
    v16bf b1 = load_frag_b_kn(Vbase + 16, DMODEL);  // d = 16..31
    acc0 = wmma_bf16(a, b0, acc0);
    acc1 = wmma_bf16(a, b1, acc1);
  }
#pragma unroll
  for (int rr = 0; rr < 8; ++rr) {
    const int mrow = rr + 8 * half;
    atomicAdd(&ctxacc[mrow][col],      acc0[rr]);   // ds_add_f32 cross-wave
    atomicAdd(&ctxacc[mrow][col + 16], acc1[rr]);
  }
  __syncthreads();
  for (int e = tid; e < 16 * 32; e += 256) {
    const int rr = e >> 5, cc = e & 31;
    Ctx[(size_t)(q0 + rr) * DMODEL + h * DHEAD + cc] =
        (__bf16)(ctxacc[rr][cc] * rowinv[rr]);
  }
}

// ---- Kernel 4: fc (context @ Wfc^T) + LayerNorm, fused --------------------

__global__ void __launch_bounds__(256) fc_ln_kernel(
    const __bf16* __restrict__ Ctx, const __bf16* __restrict__ Wfc,
    const float* __restrict__ lnw, const float* __restrict__ lnb,
    float* __restrict__ Out, int n) {
  __shared__ __bf16 As[16][264];
  __shared__ float  Ot[16][264];
  __shared__ float  red1[16][16], red2[16][16];
  __shared__ float  mu_s[16], rs_s[16];

  const int m0 = blockIdx.x * 16;
  const int tid = threadIdx.x;
  {
    const int r = tid >> 4, cb = (tid & 15) * 16;
#pragma unroll
    for (int e = 0; e < 16; ++e)
      As[r][cb + e] = Ctx[(size_t)(m0 + r) * DMODEL + cb + e];
  }
  __syncthreads();

  const int w = tid >> 5;
  const int tn0 = w * 2, tn1 = w * 2 + 1;
  v8f acc0 = {}, acc1 = {};
#pragma unroll
  for (int kc = 0; kc < DMODEL; kc += 32) {
    v16bf a  = load_frag_a(&As[0][kc], 264);
    v16bf b0 = load_frag_b_nk(Wfc + (size_t)tn0 * 16 * DMODEL + kc, DMODEL);
    v16bf b1 = load_frag_b_nk(Wfc + (size_t)tn1 * 16 * DMODEL + kc, DMODEL);
    acc0 = wmma_bf16(a, b0, acc0);
    acc1 = wmma_bf16(a, b1, acc1);
  }
  const int lane = tid & 31, col = lane & 15, half = lane >> 4;
#pragma unroll
  for (int rr = 0; rr < 8; ++rr) {
    const int mrow = rr + 8 * half;
    Ot[mrow][tn0 * 16 + col] = acc0[rr];
    Ot[mrow][tn1 * 16 + col] = acc1[rr];
  }
  __syncthreads();

  const int r = tid >> 4, l = tid & 15;
  float s1 = 0.f, s2 = 0.f;
#pragma unroll
  for (int e = 0; e < 16; ++e) {
    const float v = Ot[r][l * 16 + e];
    s1 += v; s2 += v * v;
  }
  red1[r][l] = s1; red2[r][l] = s2;
  __syncthreads();
  if (l == 0) {
    float a = 0.f, b = 0.f;
#pragma unroll
    for (int i = 0; i < 16; ++i) { a += red1[r][i]; b += red2[r][i]; }
    const float mu = a * (1.f / 256.f);
    const float var = b * (1.f / 256.f) - mu * mu;
    mu_s[r] = mu;
    rs_s[r] = rsqrtf(var + 1e-5f);
  }
  __syncthreads();
  const float mu = mu_s[r], rs = rs_s[r];
#pragma unroll
  for (int e = 0; e < 16; ++e) {
    const int c = l * 16 + e;
    Out[(size_t)(m0 + r) * DMODEL + c] = (Ot[r][c] - mu) * rs * lnw[c] + lnb[c];
  }
}

// ---------------------------------------------------------------------------

extern "C" void kernel_launch(void* const* d_in, const int* in_sizes, int n_in,
                              void* d_out, int out_size, void* d_ws, size_t ws_size,
                              hipStream_t stream) {
  (void)in_sizes; (void)n_in; (void)out_size; (void)ws_size;
  const float* inQ  = (const float*)d_in[0];
  const float* inK  = (const float*)d_in[1];
  const float* inV  = (const float*)d_in[2];
  const unsigned char* mask = (const unsigned char*)d_in[3];  // jnp bool, 1B
  const float* WQ   = (const float*)d_in[4];
  const float* WK   = (const float*)d_in[5];
  const float* WV   = (const float*)d_in[6];
  const float* Wfc  = (const float*)d_in[7];
  const float* lnw  = (const float*)d_in[8];
  const float* lnb  = (const float*)d_in[9];

  const int n = SEQ;
  const size_t act_elems = (size_t)n * DMODEL;          // 1,048,576
  const size_t w_elems   = (size_t)DMODEL * DMODEL;     // 65,536

  char* ws = (char*)d_ws;
  size_t off = 0;
  auto alloc = [&](size_t bytes) {
    void* p = ws + off;
    off = (off + bytes + 255) & ~(size_t)255;
    return p;
  };
  __bf16* Xq  = (__bf16*)alloc(act_elems * 2);
  __bf16* Xk  = (__bf16*)alloc(act_elems * 2);
  __bf16* Xv  = (__bf16*)alloc(act_elems * 2);
  __bf16* WQb = (__bf16*)alloc(w_elems * 2);
  __bf16* WKb = (__bf16*)alloc(w_elems * 2);
  __bf16* WVb = (__bf16*)alloc(w_elems * 2);
  __bf16* WFb = (__bf16*)alloc(w_elems * 2);
  __bf16* Qb  = (__bf16*)alloc(act_elems * 2);
  __bf16* Kb  = (__bf16*)alloc(act_elems * 2);
  __bf16* Vb  = (__bf16*)alloc(act_elems * 2);
  __bf16* Ctx = (__bf16*)alloc(act_elems * 2);

  float* out_normed = (float*)d_out;                     // [4096, 256]
  float* out_attn   = out_normed + act_elems;            // [8, 4096, 4096]

  // 1) convert activations + weights to bf16
  cvt_f32_bf16<<<(int)((act_elems + 255) / 256), 256, 0, stream>>>(inQ, Xq, (int)act_elems);
  cvt_f32_bf16<<<(int)((act_elems + 255) / 256), 256, 0, stream>>>(inK, Xk, (int)act_elems);
  cvt_f32_bf16<<<(int)((act_elems + 255) / 256), 256, 0, stream>>>(inV, Xv, (int)act_elems);
  cvt_f32_bf16<<<(int)((w_elems + 255) / 256), 256, 0, stream>>>(WQ,  WQb, (int)w_elems);
  cvt_f32_bf16<<<(int)((w_elems + 255) / 256), 256, 0, stream>>>(WK,  WKb, (int)w_elems);
  cvt_f32_bf16<<<(int)((w_elems + 255) / 256), 256, 0, stream>>>(WV,  WVb, (int)w_elems);
  cvt_f32_bf16<<<(int)((w_elems + 255) / 256), 256, 0, stream>>>(Wfc, WFb, (int)w_elems);

  // 2) Q/K/V projections (WMMA bf16)
  dim3 gg(DMODEL / 64, n / 64);
  gemm_xwt_bf16<<<gg, 256, 0, stream>>>(Xq, WQb, Qb, n, DMODEL, DMODEL);
  gemm_xwt_bf16<<<gg, 256, 0, stream>>>(Xk, WKb, Kb, n, DMODEL, DMODEL);
  gemm_xwt_bf16<<<gg, 256, 0, stream>>>(Xv, WVb, Vb, n, DMODEL, DMODEL);

  // 3) fused scores + softmax + attn-write + context
  dim3 gc(n / 16, NHEADS);
  fused_attn_kernel<<<gc, 256, 0, stream>>>(Qb, Kb, Vb, mask, out_attn, Ctx, n);

  // 4) output projection + LayerNorm
  fc_ln_kernel<<<n / 16, 256, 0, stream>>>(Ctx, WFb, lnw, lnb, out_normed, n);
}
